// SelectiveMagnoViT_75806172774842
// MI455X (gfx1250) — matrix operand order repaired and checked by reading, not compiled, not run on hardware
//
#include <hip/hip_runtime.h>
#include <hip/hip_bf16.h>

// ---------------------------------------------------------------------------
// SelectiveMagnoViT forward for MI455X (gfx1250, wave32, WMMA).
// All GEMMs run on v_wmma_f32_16x16x32_f16 with f16 inputs / f32 accumulate.
// Weights are converted once per call to f16, TRANSPOSED to (N,K) so the
// WMMA B operand is two contiguous 16-byte loads per lane.
// Each GEMM wave owns a 16x64 C strip: one A tile feeds 4 WMMAs.
// ---------------------------------------------------------------------------

typedef _Float16 f16;
typedef __attribute__((ext_vector_type(16))) _Float16 v16h;
typedef __attribute__((ext_vector_type(8)))  float    v8f;
typedef __attribute__((ext_vector_type(4)))  float    v4f;

#define B_N    64
#define IMG_   384
#define P_     16
#define G_     24
#define NPAT   576
#define KSEL   144
#define S_     145
#define D_     192
#define NH_    3
#define HD_    64
#define DEPTH_ 12
#define HID_   768
#define NCLS_  1000
#define NTOK   (B_N * S_)   // 9280 tokens, multiple of 16

union U16 { v4f f[2]; f16 h[16]; v16h v; };

__device__ __forceinline__ v8f wmma_f16(const U16& a, const U16& b, v8f c) {
  return __builtin_amdgcn_wmma_f32_16x16x32_f16(false, a.v, false, b.v,
                                                (short)0, c, false, false);
}

// ---------------------------------------------------------------------------
// Weight convert + transpose: src (L, K, N) f32 -> dst (L, N, K) f16
// ---------------------------------------------------------------------------
__global__ void k_tconv(const float* __restrict__ src, f16* __restrict__ dst,
                        int L, int K, int N) {
  size_t total = (size_t)L * K * N;
  size_t stride = (size_t)gridDim.x * blockDim.x;
  for (size_t i = (size_t)blockIdx.x * blockDim.x + threadIdx.x; i < total; i += stride) {
    size_t l = i / ((size_t)K * N);
    size_t r = i % ((size_t)K * N);
    size_t n = r / K;
    size_t k = r % K;
    dst[i] = (f16)src[l * (size_t)K * N + k * (size_t)N + n];
  }
}

// ---------------------------------------------------------------------------
// Patch scores (16x16 sums of line_drawing) + iterative top-144 argmax.
// One block per image; ties -> lowest index (matches jax.lax.top_k).
// ---------------------------------------------------------------------------
__global__ void k_topk(const float* __restrict__ line, int* __restrict__ idx) {
  int b = blockIdx.x, t = threadIdx.x;
  __shared__ float sv[NPAT];
  __shared__ float rv[256];
  __shared__ int   ri[256];
  for (int p = t; p < NPAT; p += 256) {
    int gy = p / G_, gx = p % G_;
    const float* base = line + ((size_t)b * IMG_ + gy * P_) * IMG_ + gx * P_;
    float s = 0.f;
    for (int iy = 0; iy < P_; ++iy)
      for (int ix = 0; ix < P_; ++ix) s += base[iy * IMG_ + ix];
    sv[p] = s;   // monotonic in the mean; divide not needed for ranking
  }
  __syncthreads();
  for (int it = 0; it < KSEL; ++it) {
    float bv = -1e30f; int bi = 0x7fffffff;
    for (int p = t; p < NPAT; p += 256) {
      float v = sv[p];
      if (v > bv || (v == bv && p < bi)) { bv = v; bi = p; }
    }
    rv[t] = bv; ri[t] = bi;
    __syncthreads();
    for (int off = 128; off > 0; off >>= 1) {
      if (t < off) {
        float v2 = rv[t + off]; int i2 = ri[t + off];
        if (v2 > rv[t] || (v2 == rv[t] && i2 < ri[t])) { rv[t] = v2; ri[t] = i2; }
      }
      __syncthreads();
    }
    if (t == 0) { idx[b * KSEL + it] = ri[0]; sv[ri[0]] = -1e30f; }
    __syncthreads();
  }
}

// ---------------------------------------------------------------------------
// Gathered patch embed: seq[b,0] = cls + pos[0]; seq[b,s] = conv(patch idx)
// + conv_b + pos[1+idx]. One block per token, 192 threads.
// ---------------------------------------------------------------------------
__global__ void k_embed(const float* __restrict__ magno,
                        const float* __restrict__ conv_w,
                        const float* __restrict__ conv_b,
                        const float* __restrict__ pos,
                        const float* __restrict__ cls,
                        const int*   __restrict__ idx,
                        float* __restrict__ seq) {
  int tkn = blockIdx.x;
  int b = tkn / S_, s = tkn % S_, d = threadIdx.x;
  if (s == 0) {
    seq[(size_t)tkn * D_ + d] = cls[d] + pos[d];
    return;
  }
  __shared__ float xp[768];
  int p = idx[b * KSEL + (s - 1)];
  int gy = p / G_, gx = p % G_;
  for (int e = d; e < 768; e += D_) {
    int c = e >> 8, rem = e & 255, iy = rem >> 4, ix = rem & 15;
    xp[e] = magno[(((size_t)b * 3 + c) * IMG_ + gy * P_ + iy) * IMG_ + gx * P_ + ix];
  }
  __syncthreads();
  float acc = conv_b[d];
  const float* wr = conv_w + (size_t)d * 768;
  for (int e = 0; e < 768; ++e) acc = fmaf(xp[e], wr[e], acc);
  seq[(size_t)tkn * D_ + d] = acc + pos[(size_t)(1 + p) * D_ + d];
}

// ---------------------------------------------------------------------------
// LayerNorm over D=192, f32 in -> f16 out. One block per token.
// row_mul lets the final LN pick token 0 of each image (row = tok * S_).
// ---------------------------------------------------------------------------
__global__ void k_ln(const float* __restrict__ in,
                     const float* __restrict__ sc,
                     const float* __restrict__ bt,
                     f16* __restrict__ out, int row_mul) {
  int tok = blockIdx.x, d = threadIdx.x;
  __shared__ float red[D_];
  __shared__ float mb, vb;
  size_t row = (size_t)tok * row_mul;
  float x = in[row * D_ + d];
  red[d] = x; __syncthreads();
  for (int off = 96; off > 1; off >>= 1) {
    if (d < off) red[d] += red[d + off];
    __syncthreads();
  }
  if (d == 0) mb = (red[0] + red[1] + red[2]) * (1.f / D_);
  __syncthreads();
  float dx = x - mb;
  red[d] = dx * dx; __syncthreads();
  for (int off = 96; off > 1; off >>= 1) {
    if (d < off) red[d] += red[d + off];
    __syncthreads();
  }
  if (d == 0) vb = (red[0] + red[1] + red[2]) * (1.f / D_);
  __syncthreads();
  float y = dx * rsqrtf(vb + 1e-6f) * sc[d] + bt[d];
  out[(size_t)tok * D_ + d] = (f16)y;
}

// ---------------------------------------------------------------------------
// WMMA GEMM: C(M,N) = A(M,K) * WT(N,K)^T + bias, optional exact GELU.
// mode 0: f16 store   mode 1: f32 += (residual add)   mode 2: f32 store.
// Block = 128 threads = 4 waves. Each wave owns a 16x64 strip of C:
// one A tile (2x16B loads) is reused by 4 WMMAs with independent
// accumulators. Out-of-range B columns read clamped (valid) memory; their
// garbage C columns are never stored (column independence of WMMA).
// ---------------------------------------------------------------------------
__global__ void k_gemm(const f16* __restrict__ A, const f16* __restrict__ WT,
                       const float* __restrict__ bias, void* __restrict__ out,
                       int M, int N, int K, int mode, int do_gelu) {
  int wave = threadIdx.x >> 5, lane = threadIdx.x & 31;
  int mtile = blockIdx.x;
  int strip = (blockIdx.y * 4 + wave) * 64;   // first col of this wave's strip
  if (strip >= N) return;                     // uniform per wave
  int lr = lane & 15, grp = lane >> 4;
  int row = mtile * 16 + lr;
  const f16* ap = A + (size_t)row * K;

  int colt[4]; const f16* bp[4];
  for (int t = 0; t < 4; ++t) {
    colt[t] = strip + t * 16 + lr;
    int cc = (colt[t] < N) ? colt[t] : (N - 1);   // clamp: stay in-bounds
    bp[t] = WT + (size_t)cc * K;
  }

  v8f c0 = {0.f,0.f,0.f,0.f,0.f,0.f,0.f,0.f};
  v8f c1 = c0, c2 = c0, c3 = c0;

  for (int kb = 0; kb < K; kb += 32) {
    U16 a;
    a.f[0] = *(const v4f*)(ap + kb + 16 * grp);       // A: K 0-7 / 16-23
    a.f[1] = *(const v4f*)(ap + kb + 16 * grp + 8);   // A: K 8-15 / 24-31
    if (kb + 32 < K) __builtin_prefetch(ap + kb + 32, 0, 1);
    U16 b0, b1, b2, b3;
    b0.f[0] = *(const v4f*)(bp[0] + kb + 16 * grp);
    b0.f[1] = *(const v4f*)(bp[0] + kb + 16 * grp + 8);
    b1.f[0] = *(const v4f*)(bp[1] + kb + 16 * grp);
    b1.f[1] = *(const v4f*)(bp[1] + kb + 16 * grp + 8);
    b2.f[0] = *(const v4f*)(bp[2] + kb + 16 * grp);
    b2.f[1] = *(const v4f*)(bp[2] + kb + 16 * grp + 8);
    b3.f[0] = *(const v4f*)(bp[3] + kb + 16 * grp);
    b3.f[1] = *(const v4f*)(bp[3] + kb + 16 * grp + 8);
    c0 = wmma_f16(a, b0, c0);
    c1 = wmma_f16(a, b1, c1);
    c2 = wmma_f16(a, b2, c2);
    c3 = wmma_f16(a, b3, c3);
  }

  for (int t = 0; t < 4; ++t) {
    v8f c = (t == 0) ? c0 : (t == 1) ? c1 : (t == 2) ? c2 : c3;
    int col = colt[t];
    if (col >= N) continue;
    float bv = bias[col];
    for (int r = 0; r < 8; ++r) {
      int orow = mtile * 16 + r + 8 * grp;   // C: row = r + 8*(lane/16)
      float v = c[r] + bv;
      if (do_gelu) v = 0.5f * v * (1.f + erff(v * 0.70710678118654752f));
      if (mode == 0)      ((f16*)out)[(size_t)orow * N + col] = (f16)v;
      else if (mode == 1) ((float*)out)[(size_t)orow * N + col] += v;
      else                ((float*)out)[(size_t)orow * N + col] = v;
    }
  }
}

// ---------------------------------------------------------------------------
// Attention for one (batch, head, 32-row chunk): S=145 padded to 160.
// Wave owns a 16-row block. Q tile is loaded ONCE (2 WMMA operands) and
// reused across all 10 score tiles; P tile (5 operands from LDS) is loaded
// once and reused across the 4 V column tiles.
// ---------------------------------------------------------------------------
__global__ void k_attn(const f16* __restrict__ qkv, f16* __restrict__ o16) {
  int chunk = blockIdx.x, h = blockIdx.y, b = blockIdx.z;
  int wave = threadIdx.x >> 5, lane = threadIdx.x & 31;
  int mtile = chunk * 2 + wave;                      // 0..9
  int lr = lane & 15, grp = lane >> 4;
  __shared__ float sc[2][16 * 160];
  __shared__ __align__(16) f16 pp[2][16 * 160];
  const int SD = 3 * D_;                             // 576
  int sm = mtile * 16 + lr; if (sm > S_ - 1) sm = S_ - 1;
  const f16* qrow = qkv + (size_t)(b * S_ + sm) * SD + h * HD_;

  // Q tile: K=64 -> two WMMA A operands, loaded once
  U16 qa0, qa1;
  qa0.f[0] = *(const v4f*)(qrow + 16 * grp);
  qa0.f[1] = *(const v4f*)(qrow + 16 * grp + 8);
  qa1.f[0] = *(const v4f*)(qrow + 32 + 16 * grp);
  qa1.f[1] = *(const v4f*)(qrow + 32 + 16 * grp + 8);

  // scores = Q K^T / sqrt(64); K-matrix consumed directly as (N,K) B operand
  for (int n = 0; n < 10; ++n) {
    int sk = n * 16 + lr; if (sk > S_ - 1) sk = S_ - 1;  // clamp; masked later
    const f16* krow = qkv + (size_t)(b * S_ + sk) * SD + D_ + h * HD_;
    U16 b0, b1;
    b0.f[0] = *(const v4f*)(krow + 16 * grp);
    b0.f[1] = *(const v4f*)(krow + 16 * grp + 8);
    b1.f[0] = *(const v4f*)(krow + 32 + 16 * grp);
    b1.f[1] = *(const v4f*)(krow + 32 + 16 * grp + 8);
    v8f c = {0.f,0.f,0.f,0.f,0.f,0.f,0.f,0.f};
    c = wmma_f16(qa0, b0, c);
    c = wmma_f16(qa1, b1, c);
    for (int r = 0; r < 8; ++r)
      sc[wave][(r + 8 * grp) * 160 + n * 16 + lr] = c[r] * 0.125f;
  }
  __syncthreads();

  // fp32 softmax over the 145 real columns; zero-pad P to 160
  if (lane < 16) {
    float* rp = sc[wave] + lane * 160;
    float m = -1e30f;
    for (int j = 0; j < S_; ++j) m = fmaxf(m, rp[j]);
    float sum = 0.f;
    for (int j = 0; j < S_; ++j) { float e = __expf(rp[j] - m); rp[j] = e; sum += e; }
    float inv = 1.f / sum;
    f16* pr = pp[wave] + lane * 160;
    for (int j = 0; j < S_; ++j) pr[j] = (f16)(rp[j] * inv);
    for (int j = S_; j < 160; ++j) pr[j] = (f16)0.f;
  }
  __syncthreads();

  // O = P V : P tile (16x160) loaded once as 5 A operands from LDS
  const f16* prow = pp[wave] + lr * 160;
  U16 pa[5];
  for (int t = 0; t < 5; ++t) {
    pa[t].f[0] = *(const v4f*)(prow + t * 32 + 16 * grp);
    pa[t].f[1] = *(const v4f*)(prow + t * 32 + 16 * grp + 8);
  }
  for (int n = 0; n < 4; ++n) {
    int dcol = 2 * D_ + h * HD_ + n * 16 + lr;
    v8f c = {0.f,0.f,0.f,0.f,0.f,0.f,0.f,0.f};
    for (int t = 0; t < 5; ++t) {
      U16 bb;
      for (int j = 0; j < 16; ++j) {
        int kk = t * 32 + 16 * grp + j;
        if (kk > S_ - 1) kk = S_ - 1;               // P=0 there, product is 0
        bb.h[j] = qkv[(size_t)(b * S_ + kk) * SD + dcol];
      }
      c = wmma_f16(pa[t], bb, c);
    }
    for (int r = 0; r < 8; ++r) {
      int so = mtile * 16 + r + 8 * grp;
      if (so < S_)
        o16[(size_t)(b * S_ + so) * D_ + h * HD_ + n * 16 + lr] = (f16)c[r];
    }
  }
}

// ---------------------------------------------------------------------------
// Host orchestration
// ---------------------------------------------------------------------------
extern "C" void kernel_launch(void* const* d_in, const int* in_sizes, int n_in,
                              void* d_out, int out_size, void* d_ws, size_t ws_size,
                              hipStream_t stream) {
  const float* magno  = (const float*)d_in[0];
  const float* line   = (const float*)d_in[1];
  const float* conv_w = (const float*)d_in[2];
  const float* conv_b = (const float*)d_in[3];
  const float* pos    = (const float*)d_in[4];
  const float* cls    = (const float*)d_in[5];
  const float* ln1s   = (const float*)d_in[6];
  const float* ln1b   = (const float*)d_in[7];
  const float* qkvw   = (const float*)d_in[8];
  const float* qkvb   = (const float*)d_in[9];
  const float* projw  = (const float*)d_in[10];
  const float* projb  = (const float*)d_in[11];
  const float* ln2s   = (const float*)d_in[12];
  const float* ln2b   = (const float*)d_in[13];
  const float* fc1w   = (const float*)d_in[14];
  const float* fc1b   = (const float*)d_in[15];
  const float* fc2w   = (const float*)d_in[16];
  const float* fc2b   = (const float*)d_in[17];
  const float* nfs    = (const float*)d_in[18];
  const float* nfb    = (const float*)d_in[19];
  const float* headw  = (const float*)d_in[20];
  const float* headb  = (const float*)d_in[21];
  float* outp = (float*)d_out;

  // bump allocator over the workspace (deterministic layout, ~50 MB)
  char* wsp = (char*)d_ws;
  auto alloc = [&](size_t bytes) -> char* {
    char* p = wsp;
    wsp += (bytes + 255) & ~(size_t)255;
    return p;
  };
  f16* wqkvT  = (f16*)alloc(sizeof(f16) * (size_t)DEPTH_ * 3 * D_ * D_);
  f16* wprojT = (f16*)alloc(sizeof(f16) * (size_t)DEPTH_ * D_ * D_);
  f16* wfc1T  = (f16*)alloc(sizeof(f16) * (size_t)DEPTH_ * HID_ * D_);
  f16* wfc2T  = (f16*)alloc(sizeof(f16) * (size_t)DEPTH_ * D_ * HID_);
  f16* wheadT = (f16*)alloc(sizeof(f16) * (size_t)NCLS_ * D_);
  f16* y16    = (f16*)alloc(sizeof(f16) * (size_t)NTOK * D_);
  f16* qkv16  = (f16*)alloc(sizeof(f16) * (size_t)NTOK * 3 * D_);
  f16* o16    = (f16*)alloc(sizeof(f16) * (size_t)NTOK * D_);
  f16* h16    = (f16*)alloc(sizeof(f16) * (size_t)NTOK * HID_);
  f16* cls16  = (f16*)alloc(sizeof(f16) * (size_t)B_N * D_);
  float* seq  = (float*)alloc(sizeof(float) * (size_t)NTOK * D_);
  int*   idx  = (int*)alloc(sizeof(int) * (size_t)B_N * KSEL);

  // weights -> f16, transposed to (N, K)
  k_tconv<<<4096, 256, 0, stream>>>(qkvw,  wqkvT,  DEPTH_, D_,   3 * D_);
  k_tconv<<<2048, 256, 0, stream>>>(projw, wprojT, DEPTH_, D_,   D_);
  k_tconv<<<4096, 256, 0, stream>>>(fc1w,  wfc1T,  DEPTH_, D_,   HID_);
  k_tconv<<<4096, 256, 0, stream>>>(fc2w,  wfc2T,  DEPTH_, HID_, D_);
  k_tconv<<<1024, 256, 0, stream>>>(headw, wheadT, 1,      D_,   NCLS_);

  // score + top-k, then gathered patch embed
  k_topk<<<B_N, 256, 0, stream>>>(line, idx);
  k_embed<<<NTOK, D_, 0, stream>>>(magno, conv_w, conv_b, pos, cls, idx, seq);

  // 12 transformer blocks  (grid.y = ceil(N / 256), 4 strips of 64 per block)
  for (int l = 0; l < DEPTH_; ++l) {
    k_ln<<<NTOK, D_, 0, stream>>>(seq, ln1s + l * D_, ln1b + l * D_, y16, 1);
    k_gemm<<<dim3(NTOK / 16, (3 * D_ + 255) / 256), 128, 0, stream>>>(
        y16, wqkvT + (size_t)l * 3 * D_ * D_, qkvb + l * 3 * D_, qkv16,
        NTOK, 3 * D_, D_, 0, 0);
    k_attn<<<dim3(5, NH_, B_N), 64, 0, stream>>>(qkv16, o16);
    k_gemm<<<dim3(NTOK / 16, (D_ + 255) / 256), 128, 0, stream>>>(
        o16, wprojT + (size_t)l * D_ * D_, projb + l * D_, seq,
        NTOK, D_, D_, 1, 0);
    k_ln<<<NTOK, D_, 0, stream>>>(seq, ln2s + l * D_, ln2b + l * D_, y16, 1);
    k_gemm<<<dim3(NTOK / 16, (HID_ + 255) / 256), 128, 0, stream>>>(
        y16, wfc1T + (size_t)l * HID_ * D_, fc1b + l * HID_, h16,
        NTOK, HID_, D_, 0, 1);
    k_gemm<<<dim3(NTOK / 16, (D_ + 255) / 256), 128, 0, stream>>>(
        h16, wfc2T + (size_t)l * D_ * HID_, fc2b + l * D_, seq,
        NTOK, D_, HID_, 1, 0);
  }

  // final LN on cls token of each image, then classifier head
  k_ln<<<B_N, D_, 0, stream>>>(seq, nfs, nfb, cls16, S_);
  k_gemm<<<dim3(B_N / 16, (NCLS_ + 255) / 256), 128, 0, stream>>>(
      cls16, wheadT, headb, outp, B_N, NCLS_, D_, 2, 0);
}
